// PrevDropout_25443386261536
// MI455X (gfx1250) — compile-verified, matrix operand order checked by reading, not run on hardware
//
#include <hip/hip_runtime.h>

typedef __attribute__((ext_vector_type(16))) __bf16 v16bf;
typedef __attribute__((ext_vector_type(8)))  float  v8f;
typedef __attribute__((ext_vector_type(4)))  float  v4f;

#define KEEP_CNT 512
#define NEG_INF  -1.0e12f

// ---------------------------------------------------------------------------
// Phase 1: mask generation. Exact JAX threefry2x32, key = jax.random.key(42)
// = (k0=0, k1=42); counts = iota(16384) split in halves. uniform = bitcast
// ((bits>>9)|0x3f800000)-1. rank<512 (stable) -> keep.
// ---------------------------------------------------------------------------
__device__ __forceinline__ unsigned rotl32(unsigned x, int n) {
  return (x << n) | (x >> (32 - n));
}

__device__ __forceinline__ void threefry2x32(unsigned k0, unsigned k1,
                                             unsigned x0, unsigned x1,
                                             unsigned& o0, unsigned& o1) {
  const unsigned ks2 = k0 ^ k1 ^ 0x1BD11BDAu;
  const int ra[4] = {13, 15, 26, 6};
  const int rb[4] = {17, 29, 16, 24};
  x0 += k0; x1 += k1;
#pragma unroll
  for (int i = 0; i < 4; ++i) { x0 += x1; x1 = rotl32(x1, ra[i]); x1 ^= x0; }
  x0 += k1; x1 += ks2 + 1u;
#pragma unroll
  for (int i = 0; i < 4; ++i) { x0 += x1; x1 = rotl32(x1, rb[i]); x1 ^= x0; }
  x0 += ks2; x1 += k0 + 2u;
#pragma unroll
  for (int i = 0; i < 4; ++i) { x0 += x1; x1 = rotl32(x1, ra[i]); x1 ^= x0; }
  x0 += k0; x1 += k1 + 3u;
#pragma unroll
  for (int i = 0; i < 4; ++i) { x0 += x1; x1 = rotl32(x1, rb[i]); x1 ^= x0; }
  x0 += k1; x1 += ks2 + 4u;
#pragma unroll
  for (int i = 0; i < 4; ++i) { x0 += x1; x1 = rotl32(x1, ra[i]); x1 ^= x0; }
  x0 += ks2; x1 += k0 + 5u;
  o0 = x0; o1 = x1;
}

__global__ __launch_bounds__(1024) void randmask_kernel(float* __restrict__ mask) {
  __shared__ float u[1024];
  const int j = threadIdx.x;
  const int b = blockIdx.x;
  const unsigned i = (unsigned)(b * 1024 + j);
  unsigned o0, o1, bits;
  if (i < 8192u) { threefry2x32(0u, 42u, i, i + 8192u, o0, o1); bits = o0; }
  else           { threefry2x32(0u, 42u, i - 8192u, i, o0, o1); bits = o1; }
  const float f = __builtin_bit_cast(float, (bits >> 9) | 0x3F800000u) - 1.0f;
  u[j] = f;
  __syncthreads();
  const float v = u[j];
  int c = 0;
  for (int k = 0; k < 1024; ++k) {
    const float w = u[k];
    c += ((w < v) || (w == v && k < j)) ? 1 : 0;  // stable argsort rank
  }
  mask[b * 1024 + j] = (c < KEEP_CNT) ? 0.0f : NEG_INF;
}

// ---------------------------------------------------------------------------
// Phase 2: out = x + broadcast(mask) via V_WMMA_F32_16X16X32_BF16.
//   D = A x B + C ; A[m,k] = bf16 expansion term k of mask(n2=base_n+m), k<3
//                   B[k,n] = 1 for k<3      -> D = C + mask broadcast over N.
// M -> n2, N -> n1-row: each lane's 8 C/D values are 8 consecutive floats,
// and one wave's load instruction covers 16 full 64-byte lines.
// ---------------------------------------------------------------------------
__device__ __forceinline__ __bf16 bf16_bits(unsigned short u) {
  return __builtin_bit_cast(__bf16, u);
}

__device__ __forceinline__ unsigned short bf16_trunc(float f, float& rem) {
  const unsigned u = __builtin_bit_cast(unsigned, f);
  const unsigned short h = (unsigned short)(u >> 16);
  rem = f - __builtin_bit_cast(float, (unsigned)h << 16);
  return h;
}

__global__ __launch_bounds__(256) void addmask_kernel(const float* __restrict__ x,
                                                      const float* __restrict__ mask,
                                                      float* __restrict__ out) {
  __shared__ float smask[1024];

  const int blk = blockIdx.x;
  const int s   = blk >> 4;   // slice = b*16 + h, 0..255
  const int sub = blk & 15;   // row-band of the slice
  const int b   = s >> 4;

  // Preload this batch's mask row (4KB) into LDS.
  {
    const float* mrow = mask + ((size_t)b << 10);
    const int t4 = threadIdx.x << 2;
    *(v4f*)&smask[t4] = *(const v4f*)&mrow[t4];
  }
  __syncthreads();

  const int w     = threadIdx.x >> 5;   // wave id 0..7
  const int lane  = threadIdx.x & 31;
  const int rlane = lane & 15;          // N (n1-row within tile)
  const int chalf = (lane >> 4) << 3;   // M offset 0 or 8 (n2)
  const bool lo16 = (lane < 16);
  const size_t slice_off = (size_t)s << 20;

  // B: ones in K=0..2 rows (lanes 0-15 hold K=0..7,16..23; elem e = K=e for e<8).
  v16bf bv;
#pragma unroll
  for (int e = 0; e < 16; ++e) bv[e] = bf16_bits(0);
  const __bf16 one = bf16_bits(0x3F80);
  bv[0] = lo16 ? one : bf16_bits(0);
  bv[1] = lo16 ? one : bf16_bits(0);
  bv[2] = lo16 ? one : bf16_bits(0);

  // Loop-invariant 3-term bf16 expansion of NEG_INF (constant-folded);
  // per tile we only select between these and zero.
  float r1, r2, r3;
  const unsigned short H0 = bf16_trunc(NEG_INF, r1);
  const unsigned short H1 = bf16_trunc(r1, r2);
  const unsigned short H2 = bf16_trunc(r2, r3);
  const __bf16 a0m = lo16 ? bf16_bits(H0) : bf16_bits(0);
  const __bf16 a1m = lo16 ? bf16_bits(H1) : bf16_bits(0);
  const __bf16 a2m = lo16 ? bf16_bits(H2) : bf16_bits(0);

  v16bf av;
#pragma unroll
  for (int e = 0; e < 16; ++e) av[e] = bf16_bits(0);

#pragma unroll 1
  for (int i = 0; i < 32; ++i) {
    const int tt = (i << 3) + w;          // 0..255 tile of this block
    const int tr = tt >> 6;               // 0..3
    const int tc = tt & 63;               // 0..63
    const int base_row = (sub << 6) + (tr << 4);  // n1 base
    const int base_n   = tc << 4;                 // n2 base

    const size_t off = slice_off + ((size_t)(base_row + rlane) << 10)
                     + (size_t)(base_n + chalf);

    // Unconditional prefetch of the next tile (wrapped in-bounds; no branch,
    // EXEC stays all-1s for the WMMA).
    {
      const int tt2 = (tt + 8) & 255;
      const int tr2 = tt2 >> 6, tc2 = tt2 & 63;
      const size_t off2 = slice_off
          + ((size_t)((sub << 6) + (tr2 << 4) + rlane) << 10)
          + (size_t)((tc2 << 4) + chalf);
      __builtin_prefetch(x + off2, 0, 1);
    }

    // A: select masked/kept expansion terms (mask value is 0 or -1e12).
    const bool masked = (smask[base_n + rlane] != 0.0f);
    av[0] = masked ? a0m : bf16_bits(0);
    av[1] = masked ? a1m : bf16_bits(0);
    av[2] = masked ? a2m : bf16_bits(0);

    // Stream 8 consecutive floats (two b128) per lane, non-temporal.
    const v8f c = __builtin_nontemporal_load((const v8f*)(x + off));
    const v8f d = __builtin_amdgcn_wmma_f32_16x16x32_bf16(
        /*neg_a=*/false, av, /*neg_b=*/false, bv,
        /*c_mod=*/(short)0, c, /*reuse_a=*/false, /*reuse_b=*/false);
    __builtin_nontemporal_store(d, (v8f*)(out + off));
  }
}

// ---------------------------------------------------------------------------
extern "C" void kernel_launch(void* const* d_in, const int* in_sizes, int n_in,
                              void* d_out, int out_size, void* d_ws, size_t ws_size,
                              hipStream_t stream) {
  (void)in_sizes; (void)n_in; (void)out_size; (void)ws_size;
  const float* x   = (const float*)d_in[0];
  float* mask      = (float*)d_ws;      // 16*1024 floats = 64 KB
  float* out       = (float*)d_out;

  randmask_kernel<<<16, 1024, 0, stream>>>(mask);
  addmask_kernel<<<4096, 256, 0, stream>>>(x, mask, out);
}
